// TiTokEncoder_30683246362814
// MI455X (gfx1250) — compile-verified, matrix operand order checked by reading, not run on hardware
//
#include <hip/hip_runtime.h>
#include <cmath>

// ---------------- model constants (from reference) ----------------
#define BB      2
#define CC      3
#define FRAMES  16
#define HWD     96
#define PT      4
#define PH      8
#define PW      8
#define GT      (FRAMES/PT)          // 4
#define GH      (HWD/PH)             // 12
#define GW      (HWD/PW)             // 12
#define NPATCH  (GT*GH*GW)           // 576
#define NTOK    128
#define SS      (NTOK+NPATCH)        // 704
#define WIDTH   512
#define HEADS   8
#define LAYERS  8
#define MLPD    2048
#define HD      (WIDTH/HEADS)        // 64
#define KIN     (CC*PT*PH*PW)        // 768
#define TOKSZ   5
#define EPSV    1e-6f

// ---------------- vector types ----------------
typedef __attribute__((ext_vector_type(16))) __bf16 v16bf;
typedef __attribute__((ext_vector_type(8)))  float  v8f;
typedef __attribute__((ext_vector_type(4)))  unsigned int u32x4;
typedef __attribute__((ext_vector_type(8)))  int   i32x8;
typedef __attribute__((ext_vector_type(4)))  int   i32x4;

__device__ __forceinline__ v8f wmma_bf16(v16bf a, v16bf b, v8f c) {
  return __builtin_amdgcn_wmma_f32_16x16x32_bf16(false, a, false, b, (short)0, c,
                                                 false, false);
}

// 16x32 bf16 fragment from LDS (row-major, leading dim LDL, bf16 elements).
// lane<16: row=lane, K chunks {0..7},{16..23}; lane>=16: row=lane-16, {8..15},{24..31}
template<int LDL>
__device__ __forceinline__ v16bf load_fragA(const __bf16* base, int lane) {
  int r  = lane & 15;
  int kh = (lane >> 4) << 3;
  const __bf16* p = base + r * LDL + kh;
  union { uint4 u[2]; v16bf v; } u;
  u.u[0] = *(const uint4*)(p);
  u.u[1] = *(const uint4*)(p + 16);
  return u.v;
}

// Same fragment but from an f32 LDS tile (TDM-loaded); convert on the fly.
template<int LDL>
__device__ __forceinline__ v16bf load_fragW(const float* base, int lane) {
  int r  = lane & 15;
  int kh = (lane >> 4) << 3;
  const float* p = base + r * LDL + kh;
  float4 f0 = *(const float4*)(p);
  float4 f1 = *(const float4*)(p + 4);
  float4 f2 = *(const float4*)(p + 16);
  float4 f3 = *(const float4*)(p + 20);
  v16bf v;
  v[0]=(__bf16)f0.x; v[1]=(__bf16)f0.y; v[2]=(__bf16)f0.z; v[3]=(__bf16)f0.w;
  v[4]=(__bf16)f1.x; v[5]=(__bf16)f1.y; v[6]=(__bf16)f1.z; v[7]=(__bf16)f1.w;
  v[8]=(__bf16)f2.x; v[9]=(__bf16)f2.y; v[10]=(__bf16)f2.z; v[11]=(__bf16)f2.w;
  v[12]=(__bf16)f3.x; v[13]=(__bf16)f3.y; v[14]=(__bf16)f3.z; v[15]=(__bf16)f3.w;
  return v;
}

__device__ __forceinline__ float wave_sum(float v) {
  #pragma unroll
  for (int o = 16; o; o >>= 1) v += __shfl_xor(v, o, 32);
  return v;
}
__device__ __forceinline__ float wave_max(float v) {
  #pragma unroll
  for (int o = 16; o; o >>= 1) v = fmaxf(v, __shfl_xor(v, o, 32));
  return v;
}

// ---------------- generic batched WMMA GEMM with TDM weight path ----------------
// C[z][m][n] = epi( alpha * sum_k A[z][m][k] * W[z][n][k] )
// A staged VGPR->bf16 LDS (double buffered); W tile DMA'd by the Tensor Data
// Mover into f32 LDS with hardware row padding (36-float stride), double buffered.
// EPI bits: 1=+bias[n], 2=GELU(exact), 4=+resid (C-indexed), 8=*alpha
template<int BM, int BN, int EPI>
__global__ __launch_bounds__((BM/32)*(BN/32)*32)
void gemm_wmma(const float* __restrict__ A, const float* __restrict__ W,
               const float* __restrict__ bias, const float* __restrict__ resid,
               float* __restrict__ C,
               int M, int N, int K, int lda, int ldw, int ldc,
               float alpha, int zInner,
               long sAo, long sAi, long sWo, long sWi, long sCo, long sCi)
{
  constexpr int NT    = (BM/32)*(BN/32)*32;
  constexpr int LDA_B = 40;          // bf16 LDS row stride for A (32 + 8 pad)
  constexpr int LDW_F = 36;          // f32  LDS row stride for W (32 + 4 TDM pad)
  const int tid  = threadIdx.x;
  const int lane = tid & 31;
  const int w    = tid >> 5;
  constexpr int WN = BN/32;
  const int wm = w / WN, wn = w % WN;
  const bool w0 = (w == 0);

  const int z     = blockIdx.z;
  const long offA = (long)(z / zInner) * sAo + (long)(z % zInner) * sAi;
  const long offW = (long)(z / zInner) * sWo + (long)(z % zInner) * sWi;
  const long offC = (long)(z / zInner) * sCo + (long)(z % zInner) * sCi;

  const int row0 = blockIdx.y * BM;
  const int col0 = blockIdx.x * BN;

  __shared__ __align__(16) __bf16 sA[2][BM * LDA_B];
  __shared__ __align__(16) float  sW[2][BN * LDW_F];

  // ---- per-thread A staging bases (compile-time trip count, fully unrolled)
  const int ar0 = tid >> 3;                 // starting row handled by this thread
  const int asg = (tid & 7) * 4;            // float4 segment within the 32-wide K tile
  const float* aBase = A + offA + (long)(row0 + ar0) * lda + asg;

  auto stage_A = [&](int buf, int k0) {
    #pragma unroll
    for (int ii = 0; ii < (BM * 8) / NT; ++ii) {
      float4 f = *(const float4*)(aBase + (long)ii * (NT / 8) * lda + k0);
      __bf16* d = &sA[buf][(ar0 + ii * (NT / 8)) * LDA_B + asg];
      d[0] = (__bf16)f.x; d[1] = (__bf16)f.y; d[2] = (__bf16)f.z; d[3] = (__bf16)f.w;
    }
  };

  // ---- TDM descriptor issue: 2-D tile, BN rows x 32 f32, row stride ldw,
  //      hardware pad: after every 32 DWORDs insert 4 DWORDs -> 36-float rows.
  const unsigned long long wByte =
      (unsigned long long)(const char*)(W + offW + (long)col0 * ldw);
  auto issue_tdm = [&](int buf, int k0) {
    unsigned long long ga = wByte + (unsigned long long)k0 * 4ull;
    unsigned ldsAddr = (unsigned)(unsigned long long)(&sW[buf][0]);
    u32x4 g0;
    g0[0] = 1u;                                   // count=1 (valid user D#)
    g0[1] = ldsAddr;                              // lds_addr
    g0[2] = (unsigned)ga;                         // global_addr[31:0]
    g0[3] = (unsigned)(ga >> 32) | 0x80000000u;   // global_addr[56:32] | type=2
    i32x8 g1;
    g1[0] = (2 << 16)      // data_size = 4 bytes
          | (1 << 20)      // pad_enable
          | (4 << 22)      // pad_interval: 32 DWORDs
          | (3 << 25);     // pad_amount: 4 DWORDs
    g1[1] = (int)(((unsigned)K & 0xFFFFu) << 16);                     // tensor_dim0 lo
    g1[2] = (int)(((unsigned)K >> 16) & 0xFFFFu)
          | (int)(((unsigned)N & 0xFFFFu) << 16);                     // dim0 hi | dim1 lo
    g1[3] = (int)(((unsigned)N >> 16) & 0xFFFFu) | (32 << 16);        // dim1 hi | tile_dim0
    g1[4] = BN;                                                       // tile_dim1 (tile_dim2=0)
    g1[5] = ldw;                                                      // tensor_dim0_stride lo
    g1[6] = 0; g1[7] = 0;
    i32x4 z4; z4[0] = 0; z4[1] = 0; z4[2] = 0; z4[3] = 0;             // groups 2/3 unused (2-D)
    i32x8 z8; z8[0]=0; z8[1]=0; z8[2]=0; z8[3]=0; z8[4]=0; z8[5]=0; z8[6]=0; z8[7]=0;
    __builtin_amdgcn_tensor_load_to_lds(g0, g1, z4, z4, z8, 0);
  };

  v8f acc[2][2];
  #pragma unroll
  for (int i = 0; i < 2; ++i)
    #pragma unroll
    for (int j = 0; j < 2; ++j)
      #pragma unroll
      for (int e = 0; e < 8; ++e) acc[i][j][e] = 0.0f;

  const int nk = K >> 5;

  // prologue: fill buffer 0
  stage_A(0, 0);
  if (w0) { issue_tdm(0, 0); __builtin_amdgcn_s_wait_tensorcnt(0); }
  __syncthreads();

  int buf = 0;
  for (int i = 0; i < nk; ++i) {
    const int k0 = i << 5;
    const int nb = buf ^ 1;
    if (i + 1 < nk) {
      stage_A(nb, k0 + 32);                      // overlaps with WMMA below
      if (w0) issue_tdm(nb, k0 + 32);            // TDM runs concurrently
      __builtin_prefetch(aBase + (long)(k0 + 64), 0, 1);  // warm K+2 of A
    }

    v16bf a0 = load_fragA<LDA_B>(&sA[buf][(wm * 32 +  0) * LDA_B], lane);
    v16bf a1 = load_fragA<LDA_B>(&sA[buf][(wm * 32 + 16) * LDA_B], lane);
    v16bf b0 = load_fragW<LDW_F>(&sW[buf][(wn * 32 +  0) * LDW_F], lane);
    v16bf b1 = load_fragW<LDW_F>(&sW[buf][(wn * 32 + 16) * LDW_F], lane);

    acc[0][0] = wmma_bf16(a0, b0, acc[0][0]);
    acc[0][1] = wmma_bf16(a0, b1, acc[0][1]);
    acc[1][0] = wmma_bf16(a1, b0, acc[1][0]);
    acc[1][1] = wmma_bf16(a1, b1, acc[1][1]);

    if (w0 && (i + 1 < nk)) __builtin_amdgcn_s_wait_tensorcnt(0);
    __syncthreads();
    buf = nb;
  }

  // epilogue; C/D layout: vgpr v -> row (+8 for upper half-wave), lane%16 -> col
  #pragma unroll
  for (int tm = 0; tm < 2; ++tm) {
    #pragma unroll
    for (int tn = 0; tn < 2; ++tn) {
      int rbase = row0 + wm * 32 + tm * 16 + ((lane >> 4) << 3);
      int c     = col0 + wn * 32 + tn * 16 + (lane & 15);
      float bv = (EPI & 1) ? bias[c] : 0.0f;
      #pragma unroll
      for (int v = 0; v < 8; ++v) {
        float x = acc[tm][tn][v];
        if (EPI & 8) x *= alpha;
        if (EPI & 1) x += bv;
        if (EPI & 2) x = 0.5f * x * (1.0f + erff(x * 0.70710678118654752f));
        long ci = offC + (long)(rbase + v) * ldc + c;
        if (EPI & 4) x += resid[ci];
        C[ci] = x;
      }
    }
  }
}

template<int BM, int BN, int EPI>
static void launch_gemm(hipStream_t st, const float* A, const float* W,
                        const float* bias, const float* resid, float* C,
                        int M, int N, int K, int lda, int ldw, int ldc,
                        float alpha, int Z, int zInner,
                        long sAo, long sAi, long sWo, long sWi, long sCo, long sCi)
{
  dim3 g(N / BN, M / BM, Z);
  dim3 b((BM/32)*(BN/32)*32);
  gemm_wmma<BM, BN, EPI><<<g, b, 0, st>>>(A, W, bias, resid, C, M, N, K, lda, ldw,
                                          ldc, alpha, zInner, sAo, sAi, sWo, sWi,
                                          sCo, sCi);
}

// ---------------- patchify: videos -> [B*NPATCH][768] ----------------
__global__ void patchify_k(const float* __restrict__ videos, float* __restrict__ pf) {
  long idx = (long)blockIdx.x * blockDim.x + threadIdx.x;
  if (idx >= (long)BB * NPATCH * KIN) return;
  int inner = (int)(idx % KIN);
  long rest = idx / KIN;
  int patch = (int)(rest % NPATCH);
  int b     = (int)(rest / NPATCH);
  int c  = inner / (PT*PH*PW);
  int r2 = inner % (PT*PH*PW);
  int pt = r2 / (PH*PW);
  int r3 = r2 % (PH*PW);
  int ph = r3 / PW, pw = r3 % PW;
  int gt = patch / (GH*GW), gh = (patch / GW) % GH, gw = patch % GW;
  int f = gt*PT + pt, y = gh*PH + ph, x = gw*PW + pw;
  pf[idx] = videos[((((long)b*CC + c)*FRAMES + f)*HWD + y)*HWD + x];
}

// ---------------- assemble x: tokens + rmsnorm(p + m) ----------------
__global__ void assemble_x(const float* __restrict__ pproj, const float* __restrict__ maskp,
                           const float* __restrict__ lnt, const float* __restrict__ lnp,
                           float* __restrict__ x) {
  int wid  = (blockIdx.x * blockDim.x + threadIdx.x) >> 5;   // row in [0, B*S)
  int lane = threadIdx.x & 31;
  int b = wid / SS, s = wid % SS;
  float m = maskp[0];
  float* xr = x + (long)wid * WIDTH;
  if (s < NTOK) {
    float r = rsqrtf(m * m + EPSV);
    #pragma unroll
    for (int i = 0; i < 16; ++i) {
      int c = lane + i * 32;
      xr[c] = m * r * lnt[c];
    }
  } else {
    const float* pr = pproj + ((long)b * NPATCH + (s - NTOK)) * WIDTH;
    float v[16]; float sum = 0.f;
    #pragma unroll
    for (int i = 0; i < 16; ++i) { v[i] = pr[lane + i * 32] + m; sum += v[i] * v[i]; }
    sum = wave_sum(sum);
    float r = rsqrtf(sum * (1.0f / WIDTH) + EPSV);
    #pragma unroll
    for (int i = 0; i < 16; ++i) { int c = lane + i * 32; xr[c] = v[i] * r * lnp[c]; }
  }
}

// ---------------- rmsnorm over rows of WIDTH ----------------
__global__ void rmsnorm_rows(const float* __restrict__ x, const float* __restrict__ w,
                             float* __restrict__ out) {
  int wid  = (blockIdx.x * blockDim.x + threadIdx.x) >> 5;
  int lane = threadIdx.x & 31;
  const float* xr = x + (long)wid * WIDTH;
  float v[16]; float s = 0.f;
  #pragma unroll
  for (int i = 0; i < 16; ++i) { v[i] = xr[lane + i * 32]; s += v[i] * v[i]; }
  s = wave_sum(s);
  float r = rsqrtf(s * (1.0f / WIDTH) + EPSV);
  float* o = out + (long)wid * WIDTH;
  #pragma unroll
  for (int i = 0; i < 16; ++i) { int c = lane + i * 32; o[c] = v[i] * r * w[c]; }
}

// ---------------- rope + qkv split; also V -> transposed [BH][HD][S] ----------------
__global__ void rope_split(const float* __restrict__ qkv, float* __restrict__ q_ro,
                           float* __restrict__ k_ro, float* __restrict__ v_t) {
  int idx = blockIdx.x * blockDim.x + threadIdx.x;      // B*S*H*32 threads
  int j = idx & 31;
  int h = (idx >> 5) & (HEADS - 1);
  int s = (idx >> 8) % SS;
  int b = (idx >> 8) / SS;
  float a = 0.0f;
  if (s >= NTOK) {
    int p = s - NTOK;
    if (j < 12)      a = (float)(p / (GH*GW))      * powf(10000.0f, -(float)j       / 12.0f);
    else if (j < 22) a = (float)((p / GW) % GH)    * powf(10000.0f, -(float)(j - 12) / 10.0f);
    else             a = (float)(p % GW)           * powf(10000.0f, -(float)(j - 22) / 10.0f);
  }
  float cs = cosf(a), sn = sinf(a);
  long base = ((long)b * SS + s) * (3 * WIDTH);
  int  off  = h * HD + 2 * j;
  float q0 = qkv[base + off],             q1 = qkv[base + off + 1];
  float k0 = qkv[base + WIDTH + off],     k1 = qkv[base + WIDTH + off + 1];
  float v0 = qkv[base + 2*WIDTH + off],   v1 = qkv[base + 2*WIDTH + off + 1];
  long bh = (long)b * HEADS + h;
  long qb = bh * (long)SS * HD + (long)s * HD + 2 * j;
  q_ro[qb]     = q0 * cs - q1 * sn;
  q_ro[qb + 1] = q0 * sn + q1 * cs;
  k_ro[qb]     = k0 * cs - k1 * sn;
  k_ro[qb + 1] = k0 * sn + k1 * cs;
  long vb = bh * (long)HD * SS;
  v_t[vb + (long)(2*j)     * SS + s] = v0;
  v_t[vb + (long)(2*j + 1) * SS + s] = v1;
}

// ---------------- softmax over rows of length S ----------------
__global__ void softmax_rows(float* __restrict__ sc) {
  int wid  = (blockIdx.x * blockDim.x + threadIdx.x) >> 5;   // row in [0, B*H*S)
  int lane = threadIdx.x & 31;
  float* r = sc + (long)wid * SS;
  float v[SS / 32]; float mx = -3.0e30f;
  #pragma unroll
  for (int i = 0; i < SS / 32; ++i) { v[i] = r[lane + i * 32]; mx = fmaxf(mx, v[i]); }
  mx = wave_max(mx);
  float sum = 0.f;
  #pragma unroll
  for (int i = 0; i < SS / 32; ++i) { v[i] = __expf(v[i] - mx); sum += v[i]; }
  sum = wave_sum(sum);
  float inv = 1.0f / sum;
  #pragma unroll
  for (int i = 0; i < SS / 32; ++i) r[lane + i * 32] = v[i] * inv;
}

// ---------------- final rmsnorm + 512->5 projection ----------------
__global__ void final_proj(const float* __restrict__ x, const float* __restrict__ lnw,
                           const float* __restrict__ pw, const float* __restrict__ pb,
                           float* __restrict__ out) {
  int wid  = (blockIdx.x * blockDim.x + threadIdx.x) >> 5;   // row in [0, B*NTOK)
  int lane = threadIdx.x & 31;
  int b = wid >> 7, t = wid & 127;
  const float* xr = x + ((long)b * SS + t) * WIDTH;
  float v[16]; float s = 0.f;
  #pragma unroll
  for (int i = 0; i < 16; ++i) { v[i] = xr[lane + i * 32]; s += v[i] * v[i]; }
  s = wave_sum(s);
  float r = rsqrtf(s * (1.0f / WIDTH) + EPSV);
  #pragma unroll
  for (int i = 0; i < 16; ++i) v[i] = v[i] * r * lnw[lane + i * 32];
  for (int o = 0; o < TOKSZ; ++o) {
    float d = 0.f;
    #pragma unroll
    for (int i = 0; i < 16; ++i) d += v[i] * pw[o * WIDTH + lane + i * 32];
    d = wave_sum(d);
    if (lane == 0) out[(long)wid * TOKSZ + o] = d + pb[o];
  }
}

// ---------------- host orchestration ----------------
extern "C" void kernel_launch(void* const* d_in, const int* in_sizes, int n_in,
                              void* d_out, int out_size, void* d_ws, size_t ws_size,
                              hipStream_t stream) {
  const float* videos     = (const float*)d_in[0];
  // d_in[1] token_counts (int) -- always NTOK, unused
  const float* mask_token = (const float*)d_in[2];
  const float* ln_pre_t_w = (const float*)d_in[3];
  const float* ln_pre_p_w = (const float*)d_in[4];
  const float* proj_in_w  = (const float*)d_in[5];
  const float* proj_in_b  = (const float*)d_in[6];
  const float* ln1_w      = (const float*)d_in[7];
  const float* qkv_w      = (const float*)d_in[8];
  const float* qkv_b      = (const float*)d_in[9];
  const float* attn_out_w = (const float*)d_in[10];
  const float* attn_out_b = (const float*)d_in[11];
  const float* ln2_w      = (const float*)d_in[12];
  const float* fc1_w      = (const float*)d_in[13];
  const float* fc1_b      = (const float*)d_in[14];
  const float* fc2_w      = (const float*)d_in[15];
  const float* fc2_b      = (const float*)d_in[16];
  const float* ln_post_w  = (const float*)d_in[17];
  const float* proj_out_w = (const float*)d_in[18];
  const float* proj_out_b = (const float*)d_in[19];
  float* out = (float*)d_out;

  // workspace carve-up (f32), ~75 MB total
  float* ws = (float*)d_ws;
  float* p_flat = ws;                    ws += (size_t)BB * NPATCH * KIN;
  float* p_proj = ws;                    ws += (size_t)BB * NPATCH * WIDTH;
  float* x      = ws;                    ws += (size_t)BB * SS * WIDTH;
  float* h      = ws;                    ws += (size_t)BB * SS * WIDTH;
  float* qkv    = ws;                    ws += (size_t)BB * SS * 3 * WIDTH;
  float* q_ro   = ws;                    ws += (size_t)BB * HEADS * SS * HD;
  float* k_ro   = ws;                    ws += (size_t)BB * HEADS * SS * HD;
  float* v_t    = ws;                    ws += (size_t)BB * HEADS * HD * SS;
  float* scores = ws;                    ws += (size_t)BB * HEADS * SS * SS;
  float* o_buf  = ws;                    ws += (size_t)BB * SS * WIDTH;
  float* m_buf  = ws;                    ws += (size_t)BB * SS * MLPD;

  const int ROWS  = BB * SS;          // 1408
  const int BH    = BB * HEADS;       // 16

  // 1) patchify
  {
    long tot = (long)BB * NPATCH * KIN;
    patchify_k<<<dim3((tot + 255) / 256), dim3(256), 0, stream>>>(videos, p_flat);
  }
  // 2) proj_in GEMM: [1152 x 768] x [512 x 768]^T -> [1152 x 512] (+bias)
  launch_gemm<128, 64, 1>(stream, p_flat, proj_in_w, proj_in_b, nullptr, p_proj,
                          BB * NPATCH, WIDTH, KIN, KIN, KIN, WIDTH,
                          1.0f, 1, 1, 0, 0, 0, 0, 0, 0);
  // 3) assemble x
  assemble_x<<<dim3(ROWS * 32 / 256), dim3(256), 0, stream>>>(p_proj, mask_token,
                                                              ln_pre_t_w, ln_pre_p_w, x);

  for (int l = 0; l < LAYERS; ++l) {
    const float* l_ln1 = ln1_w + (long)l * WIDTH;
    const float* l_qw  = qkv_w + (long)l * 3 * WIDTH * WIDTH;
    const float* l_qb  = qkv_b + (long)l * 3 * WIDTH;
    const float* l_ow  = attn_out_w + (long)l * WIDTH * WIDTH;
    const float* l_ob  = attn_out_b + (long)l * WIDTH;
    const float* l_ln2 = ln2_w + (long)l * WIDTH;
    const float* l_f1w = fc1_w + (long)l * MLPD * WIDTH;
    const float* l_f1b = fc1_b + (long)l * MLPD;
    const float* l_f2w = fc2_w + (long)l * WIDTH * MLPD;
    const float* l_f2b = fc2_b + (long)l * WIDTH;

    rmsnorm_rows<<<dim3(ROWS * 32 / 256), dim3(256), 0, stream>>>(x, l_ln1, h);
    // qkv = h @ qkv_w^T + b
    launch_gemm<128, 64, 1>(stream, h, l_qw, l_qb, nullptr, qkv,
                            ROWS, 3 * WIDTH, WIDTH, WIDTH, WIDTH, 3 * WIDTH,
                            1.0f, 1, 1, 0, 0, 0, 0, 0, 0);
    rope_split<<<dim3(BB * SS * HEADS * (HD / 2) / 256), dim3(256), 0, stream>>>(
        qkv, q_ro, k_ro, v_t);
    // scores[z] = (Q @ K^T) / sqrt(HD), z = b*H+h
    launch_gemm<64, 64, 8>(stream, q_ro, k_ro, nullptr, nullptr, scores,
                           SS, SS, HD, HD, HD, SS,
                           0.125f, BH, 1,
                           (long)SS * HD, 0, (long)SS * HD, 0, (long)SS * SS, 0);
    softmax_rows<<<dim3(BH * SS * 32 / 256), dim3(256), 0, stream>>>(scores);
    // o[b][q][h*HD + n] = P @ V ; W operand = V^T (HD x S)
    launch_gemm<64, 64, 0>(stream, scores, v_t, nullptr, nullptr, o_buf,
                           SS, HD, SS, SS, SS, WIDTH,
                           1.0f, BH, HEADS,
                           (long)HEADS * SS * SS, (long)SS * SS,
                           (long)HEADS * HD * SS, (long)HD * SS,
                           (long)SS * WIDTH, (long)HD);
    // x = x + o @ attn_out_w^T + b
    launch_gemm<128, 64, 5>(stream, o_buf, l_ow, l_ob, x, x,
                            ROWS, WIDTH, WIDTH, WIDTH, WIDTH, WIDTH,
                            1.0f, 1, 1, 0, 0, 0, 0, 0, 0);
    rmsnorm_rows<<<dim3(ROWS * 32 / 256), dim3(256), 0, stream>>>(x, l_ln2, h);
    // m = gelu(h @ fc1^T + b)
    launch_gemm<128, 64, 3>(stream, h, l_f1w, l_f1b, nullptr, m_buf,
                            ROWS, MLPD, WIDTH, WIDTH, WIDTH, MLPD,
                            1.0f, 1, 1, 0, 0, 0, 0, 0, 0);
    // x = x + m @ fc2^T + b
    launch_gemm<128, 64, 5>(stream, m_buf, l_f2w, l_f2b, x, x,
                            ROWS, WIDTH, MLPD, MLPD, MLPD, WIDTH,
                            1.0f, 1, 1, 0, 0, 0, 0, 0, 0);
  }

  final_proj<<<dim3(BB * NTOK * 32 / 256), dim3(256), 0, stream>>>(
      x, ln_post_w, proj_out_w, proj_out_b, out);
}
